// EGNN_79044578115826
// MI455X (gfx1250) — matrix operand order repaired and checked by reading, not compiled, not run on hardware
//
#include <hip/hip_runtime.h>
#include <hip/hip_bf16.h>
#include <math.h>

// Problem constants (match reference)
#define Bq   4
#define Nq   10000
#define Eq   160000
#define Dq   128
#define OUTq 64
#define Lq   4
#define HPAD 136   // padded LDS row (ushort) -> conflict-free b128 reads

typedef __attribute__((ext_vector_type(16))) __bf16 v16bf;
typedef __attribute__((ext_vector_type(8)))  __bf16 v8bf;
typedef __attribute__((ext_vector_type(8)))  float  v8f;

#define CAT16(lo,hi) __builtin_shufflevector(lo,hi,0,1,2,3,4,5,6,7,8,9,10,11,12,13,14,15)

__device__ __forceinline__ unsigned short f2bf(float f) {
  unsigned int u = __float_as_uint(f);
  u += 0x7FFFu + ((u >> 16) & 1u);          // round-to-nearest-even
  return (unsigned short)(u >> 16);
}
__device__ __forceinline__ float lrelu(float x) { return x > 0.f ? x : 0.01f * x; }

// Load the 8 B-fragments (one per 16-col tile) of K-block kb into registers.
// Packed layout: ((kb*8+nt)*32+lane)*16 + e   (B-matrix layout, ISA 7.12.2).
__device__ __forceinline__ void load_bfrags(const unsigned short* __restrict__ wp,
                                            int kb, int lane, v8bf* lo, v8bf* hi) {
#pragma unroll
  for (int nt = 0; nt < 8; ++nt) {
    const unsigned short* bp = wp + (((kb * 8 + nt) * 32 + lane) << 4);
    lo[nt] = *(const v8bf*)bp;
    hi[nt] = *(const v8bf*)(bp + 8);
  }
}

// 8 back-to-back WMMAs: one A-fragment against 8 resident B-fragments.
__device__ __forceinline__ void wmma8(v8f acc[8], v16bf A, const v8bf* lo, const v8bf* hi) {
#pragma unroll
  for (int nt = 0; nt < 8; ++nt)
    acc[nt] = __builtin_amdgcn_wmma_f32_16x16x32_bf16(
        false, A, false, CAT16(lo[nt], hi[nt]), (short)0, acc[nt], false, false);
}

__device__ __forceinline__ v16bf load_afrag(const unsigned short* __restrict__ ap) {
  v8bf lo = *(const v8bf*)ap;
  v8bf hi = *(const v8bf*)(ap + 16);
  return CAT16(lo, hi);
}

// ---------------------------------------------------------------------------
// Weight packing: fp32 [L][K][128] row-major -> bf16 B-fragment layout.
// ---------------------------------------------------------------------------
__global__ void pack_w(const float* __restrict__ src, unsigned short* __restrict__ dst,
                       int K, int srcLayerStride, int dstLayerStride) {
  int layer = blockIdx.y;
  int tid = blockIdx.x * 256 + threadIdx.x;
  if (tid >= K * Dq) return;
  int e    = tid & 15;
  int lane = (tid >> 4) & 31;
  int nt   = (tid >> 9) & 7;
  int kb   = tid >> 12;
  int kin  = (lane >> 4) * 8 + e + ((e >> 3) << 3);
  int col  = nt * 16 + (lane & 15);
  float v  = src[(size_t)layer * srcLayerStride + (size_t)(kb * 32 + kin) * Dq + col];
  dst[(size_t)layer * dstLayerStride + tid] = f2bf(v);
}

__global__ void zero_f(float* __restrict__ p, size_t n) {
  size_t t = (size_t)blockIdx.x * blockDim.x + threadIdx.x;
  if (t < n) p[t] = 0.f;
}

__global__ void count_deg(const int* __restrict__ col, float* __restrict__ cnt, int E) {
  int t = blockIdx.x * 256 + threadIdx.x;
  if (t < E) atomicAdd(&cnt[col[t]], 1.f);
}

// h = lrelu(pos @ w0 + b0); also copy pos into workspace.
__global__ void init_h(const float* __restrict__ x, const float* __restrict__ w0,
                       const float* __restrict__ b0, float* __restrict__ hf,
                       unsigned short* __restrict__ hb, float* __restrict__ posw) {
  size_t t = (size_t)blockIdx.x * 256 + threadIdx.x;
  if (t >= (size_t)Bq * Nq * Dq) return;
  int d = (int)(t & (Dq - 1));
  size_t bn = t >> 7;                     // b*N + n
  const float* p = x + bn * 3;
  float v = b0[d] + p[0] * w0[d] + p[1] * w0[Dq + d] + p[2] * w0[2 * Dq + d];
  v = lrelu(v);
  hf[t] = v;
  hb[t] = f2bf(v);
  if (d < 3) posw[bn * 3 + d] = p[d];
}

// ---------------------------------------------------------------------------
// Edge kernel: one wave per 16-edge tile.
//   edge_in = [h[row] | h[col] | dist2]  (K = 257, last row folded into acc init)
//   msg = lrelu(lrelu(edge_in@ew1+eb1)@ew2+eb2) ; atomic scatter-add into aggr.
// GEMMs are software-pipelined: fragments of K-block kb+1 are fetched while
// the 8 WMMAs of block kb execute (register double buffering).
// ---------------------------------------------------------------------------
__global__ __launch_bounds__(256) void egnn_edge(
    const unsigned short* __restrict__ hb, const float* __restrict__ posw,
    const int* __restrict__ row, const int* __restrict__ colv,
    const unsigned short* __restrict__ ew1p, const unsigned short* __restrict__ ew2p,
    const float* __restrict__ ew1last,
    const float* __restrict__ eb1, const float* __restrict__ eb2,
    float* __restrict__ aggr) {
  __shared__ unsigned short sh[8 * 16 * HPAD];
  __shared__ float sdsq[8 * 16];
  __shared__ int   scol[8 * 16];
  const int wave = threadIdx.x >> 5, lane = threadIdx.x & 31;
  const int b = blockIdx.y;
  const int tile = blockIdx.x * 8 + wave;
  if (tile >= Eq / 16) return;
  const int m = lane & 15, hi = lane >> 4, kbase = hi * 8;
  unsigned short* tS = sh + wave * 16 * HPAD;
  float* dsq = sdsq + wave * 16;
  int*   ct  = scol + wave * 16;

  const int e  = tile * 16 + m;
  const int ri = row[e], ci = colv[e];
  if (hi == 0) {                                    // lanes 0-15: dist2 + col idx
    const float* pr = posw + ((size_t)b * Nq + ri) * 3;
    const float* pc = posw + ((size_t)b * Nq + ci) * 3;
    float dx = pr[0] - pc[0], dy = pr[1] - pc[1], dz = pr[2] - pc[2];
    dsq[m] = dx * dx + dy * dy + dz * dz;
    ct[m]  = ci;
  }
  float dq[8]; int cm[8];                           // same-wave DS is in-order
#pragma unroll
  for (int r = 0; r < 8; ++r) { dq[r] = dsq[r + kbase]; cm[r] = ct[r + kbase]; }

  const unsigned short* hrow = hb + ((size_t)b * Nq + ri) * Dq;
  const unsigned short* hcol = hb + ((size_t)b * Nq + ci) * Dq;

  // Prime pipeline: A/B fragments for kb = 0.
  v8bf bl[2][8], bhf[2][8];
  v16bf Ar[2];
  Ar[0] = load_afrag(hrow + kbase);
  load_bfrags(ew1p, 0, lane, bl[0], bhf[0]);

  // GEMM1 acc init: bias + dist2 * ew1[256,:]  (hides the priming loads)
  v8f acc[8];
#pragma unroll
  for (int nt = 0; nt < 8; ++nt) {
    const int n = nt * 16 + m;
    const float bias = eb1[n], w2 = ew1last[n];
#pragma unroll
    for (int r = 0; r < 8; ++r) acc[nt][r] = __builtin_fmaf(dq[r], w2, bias);
  }

#pragma unroll
  for (int kb = 0; kb < 8; ++kb) {                  // K = 256 over bf16 WMMA
    const int cur = kb & 1, nxt = cur ^ 1;
    if (kb < 7) {
      const int k1 = kb + 1;
      Ar[nxt] = load_afrag(((k1 < 4) ? hrow : hcol) + (k1 & 3) * 32 + kbase);
      load_bfrags(ew1p, k1, lane, bl[nxt], bhf[nxt]);
    }
    wmma8(acc, Ar[cur], bl[cur], bhf[cur]);
  }

  // Prefetch GEMM2's first weight block before the LDS round-trip.
  load_bfrags(ew2p, 0, lane, bl[0], bhf[0]);

  // lrelu -> bf16 hidden tile in LDS (A-layout reload below)
#pragma unroll
  for (int nt = 0; nt < 8; ++nt)
#pragma unroll
    for (int r = 0; r < 8; ++r)
      tS[(r + kbase) * HPAD + nt * 16 + m] = f2bf(lrelu(acc[nt][r]));

  // GEMM2: msg = hidden @ ew2 + eb2
#pragma unroll
  for (int nt = 0; nt < 8; ++nt) {
    const float bias = eb2[nt * 16 + m];
#pragma unroll
    for (int r = 0; r < 8; ++r) acc[nt][r] = bias;
  }
  Ar[0] = load_afrag(tS + m * HPAD + kbase);
#pragma unroll
  for (int kb = 0; kb < 4; ++kb) {
    const int cur = kb & 1, nxt = cur ^ 1;
    if (kb < 3) {
      Ar[nxt] = load_afrag(tS + m * HPAD + (kb + 1) * 32 + kbase);
      load_bfrags(ew2p, kb + 1, lane, bl[nxt], bhf[nxt]);
    }
    wmma8(acc, Ar[cur], bl[cur], bhf[cur]);
  }

  // scatter-mean numerator: atomic fp32 adds into aggr[b][col][:]
  float* ab = aggr + (size_t)b * Nq * Dq;
#pragma unroll
  for (int nt = 0; nt < 8; ++nt) {
    const int n = nt * 16 + m;
#pragma unroll
    for (int r = 0; r < 8; ++r)
      atomicAdd(ab + (size_t)cm[r] * Dq + n, lrelu(acc[nt][r]));
  }
}

// ---------------------------------------------------------------------------
// Node kernel: one wave per 16-node tile.
//   aggr_mean -> coord update (tanh) -> pos ;  h += MLP([h|aggr_mean])
// ---------------------------------------------------------------------------
__global__ __launch_bounds__(256) void egnn_node(
    unsigned short* __restrict__ hb, float* __restrict__ hf,
    float* __restrict__ posw, const float* __restrict__ aggr,
    const float* __restrict__ cnt,
    const unsigned short* __restrict__ nw1p, const unsigned short* __restrict__ nw2p,
    const float* __restrict__ nb1, const float* __restrict__ nb2,
    const float* __restrict__ cw, const float* __restrict__ cb) {
  __shared__ unsigned short sh[8 * 16 * HPAD];
  __shared__ float sdot[8 * 32];
  const int wave = threadIdx.x >> 5, lane = threadIdx.x & 31;
  const int tile = blockIdx.x * 8 + wave;
  if (tile >= Nq / 16) return;
  const int b = blockIdx.y;
  const int m = lane & 15, hi = lane >> 4, kbase = hi * 8;
  unsigned short* tS = sh + wave * 16 * HPAD;
  float* sd = sdot + wave * 32;
  const int node = tile * 16 + m;

  // Phase 1: aggr_mean -> bf16 LDS tile + partial dot with cw (lane handles 64 k's)
  const float* ar = aggr + ((size_t)b * Nq + node) * Dq;
  const float invd = 1.f / fmaxf(cnt[node], 1.f);
  float dot = 0.f;
  const int k0 = hi * 64;
#pragma unroll
  for (int k = 0; k < 64; k += 4) {
    float4 a4 = *(const float4*)(ar + k0 + k);
    a4.x *= invd; a4.y *= invd; a4.z *= invd; a4.w *= invd;
    dot += a4.x * cw[k0 + k] + a4.y * cw[k0 + k + 1] +
           a4.z * cw[k0 + k + 2] + a4.w * cw[k0 + k + 3];
    tS[m * HPAD + k0 + k + 0] = f2bf(a4.x);
    tS[m * HPAD + k0 + k + 1] = f2bf(a4.y);
    tS[m * HPAD + k0 + k + 2] = f2bf(a4.z);
    tS[m * HPAD + k0 + k + 3] = f2bf(a4.w);
  }
  sd[lane] = dot;
  if (hi == 0) {                                    // coordinate update
    float s = sd[m] + sd[m + 16] + cb[0];
    float cu = tanhf(s) * 0.1f;
    float* pp = posw + ((size_t)b * Nq + node) * 3;
    pp[0] += cu; pp[1] += cu; pp[2] += cu;
  }

  // GEMM1: [h | aggr_mean] @ nw1 + nb1   (kb 0-3 from global h_bf16, 4-7 from LDS)
  const unsigned short* hrow = hb + ((size_t)b * Nq + node) * Dq;
  v8bf bl[2][8], bhf[2][8];
  v16bf Ar2[2];
  Ar2[0] = load_afrag(hrow + kbase);
  load_bfrags(nw1p, 0, lane, bl[0], bhf[0]);

  v8f acc[8];
#pragma unroll
  for (int nt = 0; nt < 8; ++nt) {
    const float bias = nb1[nt * 16 + m];
#pragma unroll
    for (int r = 0; r < 8; ++r) acc[nt][r] = bias;
  }
#pragma unroll
  for (int kb = 0; kb < 8; ++kb) {
    const int cur = kb & 1, nxt = cur ^ 1;
    if (kb < 7) {
      const int k1 = kb + 1;
      Ar2[nxt] = (k1 < 4)
          ? load_afrag(hrow + k1 * 32 + kbase)
          : load_afrag(tS + m * HPAD + (k1 - 4) * 32 + kbase);
      load_bfrags(nw1p, k1, lane, bl[nxt], bhf[nxt]);
    }
    wmma8(acc, Ar2[cur], bl[cur], bhf[cur]);
  }

  // Prefetch GEMM2's first weight block before the LDS round-trip.
  load_bfrags(nw2p, 0, lane, bl[0], bhf[0]);

  // lrelu -> hidden tile (reuse same LDS; same-wave DS ordering makes this safe)
#pragma unroll
  for (int nt = 0; nt < 8; ++nt)
#pragma unroll
    for (int r = 0; r < 8; ++r)
      tS[(r + kbase) * HPAD + nt * 16 + m] = f2bf(lrelu(acc[nt][r]));

  // GEMM2: hidden @ nw2 + nb2
#pragma unroll
  for (int nt = 0; nt < 8; ++nt) {
    const float bias = nb2[nt * 16 + m];
#pragma unroll
    for (int r = 0; r < 8; ++r) acc[nt][r] = bias;
  }
  Ar2[0] = load_afrag(tS + m * HPAD + kbase);
#pragma unroll
  for (int kb = 0; kb < 4; ++kb) {
    const int cur = kb & 1, nxt = cur ^ 1;
    if (kb < 3) {
      Ar2[nxt] = load_afrag(tS + m * HPAD + (kb + 1) * 32 + kbase);
      load_bfrags(nw2p, kb + 1, lane, bl[nxt], bhf[nxt]);
    }
    wmma8(acc, Ar2[cur], bl[cur], bhf[cur]);
  }

  // Residual update of fp32 master h + bf16 shadow
  float* hfb = hf + (size_t)b * Nq * Dq;
  unsigned short* hbb = hb + (size_t)b * Nq * Dq;
#pragma unroll
  for (int nt = 0; nt < 8; ++nt) {
    const int n = nt * 16 + m;
#pragma unroll
    for (int r = 0; r < 8; ++r) {
      size_t off = (size_t)(tile * 16 + r + kbase) * Dq + n;
      float nh = hfb[off] + lrelu(acc[nt][r]);
      hfb[off] = nh;
      hbb[off] = f2bf(nh);
    }
  }
}

// Mean over nodes: hmean[b][d] = sum_n h[b][n][d] / N
__global__ void reduce_mean(const float* __restrict__ hf, float* __restrict__ hmean) {
  const int b = blockIdx.y, d = blockIdx.x;
  __shared__ float s[256];
  float acc = 0.f;
  for (int n = threadIdx.x; n < Nq; n += 256)
    acc += hf[((size_t)b * Nq + n) * Dq + d];
  s[threadIdx.x] = acc;
  __syncthreads();
  for (int st = 128; st > 0; st >>= 1) {
    if (threadIdx.x < st) s[threadIdx.x] += s[threadIdx.x + st];
    __syncthreads();
  }
  if (threadIdx.x == 0) hmean[b * Dq + d] = s[0] / (float)Nq;
}

// out = lrelu(hmean @ wp + bp)   (B*OUT = 256 threads)
__global__ void project_out(const float* __restrict__ hmean, const float* __restrict__ wp,
                            const float* __restrict__ bp, float* __restrict__ out) {
  const int t = threadIdx.x;
  const int b = t >> 6, o = t & (OUTq - 1);
  float acc = bp[o];
  for (int d = 0; d < Dq; ++d) acc += hmean[b * Dq + d] * wp[d * OUTq + o];
  out[t] = lrelu(acc);
}

// ---------------------------------------------------------------------------
extern "C" void kernel_launch(void* const* d_in, const int* in_sizes, int n_in,
                              void* d_out, int out_size, void* d_ws, size_t ws_size,
                              hipStream_t stream) {
  (void)in_sizes; (void)n_in; (void)out_size; (void)ws_size;
  const float* x   = (const float*)d_in[0];
  const int*   ei  = (const int*)d_in[1];
  const float* w0  = (const float*)d_in[2];
  const float* b0  = (const float*)d_in[3];
  const float* ew1 = (const float*)d_in[4];
  const float* eb1 = (const float*)d_in[5];
  const float* ew2 = (const float*)d_in[6];
  const float* eb2 = (const float*)d_in[7];
  const float* cw  = (const float*)d_in[8];
  const float* cb  = (const float*)d_in[9];
  const float* nw1 = (const float*)d_in[10];
  const float* nb1 = (const float*)d_in[11];
  const float* nw2 = (const float*)d_in[12];
  const float* nb2 = (const float*)d_in[13];
  const float* wp  = (const float*)d_in[14];
  const float* bp  = (const float*)d_in[15];

  char* ws = (char*)d_ws;
  size_t off = 0;
  auto carve = [&](size_t bytes) -> void* {
    void* p = ws + off;
    off += (bytes + 255) & ~(size_t)255;
    return p;
  };
  float*          posw  = (float*)carve((size_t)Bq * Nq * 3 * sizeof(float));
  float*          hf    = (float*)carve((size_t)Bq * Nq * Dq * sizeof(float));
  unsigned short* hbuf  = (unsigned short*)carve((size_t)Bq * Nq * Dq * sizeof(unsigned short));
  float*          aggr  = (float*)carve((size_t)Bq * Nq * Dq * sizeof(float));
  float*          cnt   = (float*)carve((size_t)Nq * sizeof(float));
  float*          hmean = (float*)carve((size_t)Bq * Dq * sizeof(float));
  unsigned short* ew1p  = (unsigned short*)carve((size_t)Lq * 256 * Dq * sizeof(unsigned short));
  unsigned short* ew2p  = (unsigned short*)carve((size_t)Lq * 128 * Dq * sizeof(unsigned short));
  unsigned short* nw1p  = (unsigned short*)carve((size_t)Lq * 256 * Dq * sizeof(unsigned short));
  unsigned short* nw2p  = (unsigned short*)carve((size_t)Lq * 128 * Dq * sizeof(unsigned short));

  const int* row = ei;
  const int* col = ei + Eq;

  // Pack weights into bf16 WMMA B-fragment layout (once per call; deterministic).
  pack_w<<<dim3(128, Lq), 256, 0, stream>>>(ew1, ew1p, 256, 257 * Dq, 256 * Dq);
  pack_w<<<dim3(64,  Lq), 256, 0, stream>>>(ew2, ew2p, 128, 128 * Dq, 128 * Dq);
  pack_w<<<dim3(128, Lq), 256, 0, stream>>>(nw1, nw1p, 256, 256 * Dq, 256 * Dq);
  pack_w<<<dim3(64,  Lq), 256, 0, stream>>>(nw2, nw2p, 128, 128 * Dq, 128 * Dq);

  // In-degree counts (shared by all layers/batches).
  zero_f<<<(Nq + 255) / 256, 256, 0, stream>>>(cnt, (size_t)Nq);
  count_deg<<<(Eq + 255) / 256, 256, 0, stream>>>(col, cnt, Eq);

  // h0 = lrelu(pos @ w0 + b0), plus pos copy into workspace.
  const size_t hTot = (size_t)Bq * Nq * Dq;
  init_h<<<(unsigned)((hTot + 255) / 256), 256, 0, stream>>>(x, w0, b0, hf, hbuf, posw);

  for (int layer = 0; layer < Lq; ++layer) {
    zero_f<<<(unsigned)((hTot + 255) / 256), 256, 0, stream>>>(aggr, hTot);
    egnn_edge<<<dim3(Eq / 16 / 8, Bq), 256, 0, stream>>>(
        hbuf, posw, row, col,
        ew1p + (size_t)layer * 256 * Dq, ew2p + (size_t)layer * 128 * Dq,
        ew1 + (size_t)layer * 257 * Dq + 256 * Dq,
        eb1 + (size_t)layer * Dq, eb2 + (size_t)layer * Dq, aggr);
    egnn_node<<<dim3((Nq / 16 + 7) / 8, Bq), 256, 0, stream>>>(
        hbuf, hf, posw, aggr, cnt,
        nw1p + (size_t)layer * 256 * Dq, nw2p + (size_t)layer * 128 * Dq,
        nb1 + (size_t)layer * Dq, nb2 + (size_t)layer * Dq,
        cw + (size_t)layer * Dq, cb + layer);
  }

  reduce_mean<<<dim3(Dq, Bq), 256, 0, stream>>>(hf, hmean);
  project_out<<<1, 256, 0, stream>>>(hmean, wp, bp, (float*)d_out);
}
